// MemoryReader_47536698032738
// MI455X (gfx1250) — matrix operand order repaired and checked by reading, not compiled, not run on hardware
//
#include <hip/hip_runtime.h>
#include <math.h>
#include <stdint.h>

// ---------------------------------------------------------------------------
// MemoryReader for MI455X (gfx1250, wave32, WMMA bf16 16x16x32)
// B=16, M=4096, D=1024, Q=64, H=16, HD=64.  SLOT_MASK is block-diagonal:
// query qi attends exactly to slots [qi*64, qi*64+64). Attention is folded
// through the K/V projections so `memory` is only contracted against 64-long
// per-head vectors -> single HBM stream of `memory`; WMMA does all GEMMs.
// Async global->LDS (GLOBAL_LOAD_ASYNC_TO_LDS_B128) moves bf16 tiles.
// ---------------------------------------------------------------------------

#define Bc 16
#define Mc 4096
#define Dc 1024
#define Qc 64
#define Hc 16
#define RQ 1024 // B*Q rows

typedef __attribute__((ext_vector_type(16))) __bf16 v16bf;
typedef __attribute__((ext_vector_type(8)))  __bf16 v8bf;
typedef __attribute__((ext_vector_type(4)))  __bf16 v4bf;
typedef __attribute__((ext_vector_type(8)))  float  v8f;

__device__ __forceinline__ v8f wmma_bf16(v16bf a, v16bf b, v8f c) {
#if defined(__HIP_DEVICE_COMPILE__)
  return __builtin_amdgcn_wmma_f32_16x16x32_bf16(false, a, false, b, (short)0, c,
                                                 false, false);
#else
  (void)a; (void)b; return c;
#endif
}

__device__ __forceinline__ v8f vzero8() {
  v8f v;
#pragma unroll
  for (int i = 0; i < 8; ++i) v[i] = 0.f;
  return v;
}

// Async copy 16B global -> LDS per lane (CDNA5 GLOBAL_LOAD_ASYNC_TO_LDS_B128,
// tracked by ASYNCcnt). LDS dest = low 32 bits of the flat shared pointer
// (LDS aperture: hardware adds LDS_BASE, ISA 10.2 / 08_async_tensor 4.4).
__device__ __forceinline__ void async_copy_b128(void* lds_ptr, const void* gptr) {
#if defined(__HIP_DEVICE_COMPILE__)
  uint32_t loff = (uint32_t)(uintptr_t)lds_ptr;
  asm volatile("global_load_async_to_lds_b128 %0, %1, off"
               :: "v"(loff), "v"(gptr)
               : "memory");
#else
  (void)lds_ptr; (void)gptr;
#endif
}

__device__ __forceinline__ void wait_async0() {
#if defined(__HIP_DEVICE_COMPILE__)
  asm volatile("s_wait_asynccnt 0x0" ::: "memory");
#endif
}

// A fragment (16x32 bf16) from LDS, row-major [rows][stride].
// lanes 0-15: row m=lane,    K = kBase+{0..15}
// lanes16-31: row m=lane-16, K = kBase+{16..31}
__device__ __forceinline__ v16bf load_a_frag(const __bf16* lds, int stride,
                                             int mBase, int kBase, int lane) {
  const int row = mBase + (lane & 15);
  const int k0  = kBase + ((lane & 16) ? 16 : 0);
  union { v16bf v; v8bf h[2]; } u;
  u.h[0] = *(const v8bf*)(lds + row * stride + k0);
  u.h[1] = *(const v8bf*)(lds + row * stride + k0 + 8);
  return u.v;
}

// B fragment (32x16 bf16): lane holds fixed N = nBase + lane%16, 16 contiguous
// K values at kGlobal + (lane<16 ? 0 : 16). Weights stored [n][k] row-major.
__device__ __forceinline__ v16bf load_b_frag(const __bf16* w, int ld,
                                             int nBase, int kGlobal, int lane) {
  const int n  = nBase + (lane & 15);
  const int k0 = kGlobal + ((lane & 16) ? 16 : 0);
  return *(const v16bf*)(w + (size_t)n * ld + k0);
}

// ---------------------------------------------------------------------------
// K0: convert weights to bf16 (qw, vw, out_proj row-major; kw transposed so B
// fragments for the kq GEMM are contiguous in the head-column dimension).
// ---------------------------------------------------------------------------
__global__ void k_wcvt(const float* __restrict__ in_w,
                       const float* __restrict__ out_w,
                       __bf16* __restrict__ qw, __bf16* __restrict__ kwT,
                       __bf16* __restrict__ vw, __bf16* __restrict__ ow) {
  int idx = blockIdx.x * 256 + threadIdx.x; // [0, 4*1024*1024)
  int which = idx >> 20;
  int rem = idx & 0xFFFFF;
  int r = rem >> 10, c = rem & 1023;
  if (which == 0)      qw[rem]  = (__bf16)in_w[rem];
  else if (which == 1) kwT[rem] = (__bf16)in_w[(size_t)(Dc + c) * Dc + r];
  else if (which == 2) vw[rem]  = (__bf16)in_w[(size_t)2 * Dc * Dc + rem];
  else                 ow[rem]  = (__bf16)out_w[rem];
}

// ---------------------------------------------------------------------------
// K1: cond = context @ ctx_w^T + ctx_b ; q = queries + cond (f32 + bf16 copy)
// ---------------------------------------------------------------------------
__global__ void k_condq(const float* __restrict__ context,
                        const float* __restrict__ ctx_w,
                        const float* __restrict__ ctx_b,
                        const float* __restrict__ queries,
                        float* __restrict__ qf, __bf16* __restrict__ qbf) {
  int gid = blockIdx.x * 256 + threadIdx.x; // [0, 16384)
  int b = gid >> 10, d = gid & 1023;
  const float* ctx  = context + (size_t)b * Dc;
  const float* wrow = ctx_w + (size_t)d * Dc;
  float acc = ctx_b[d];
  for (int k = 0; k < Dc; k += 4) {
    acc = fmaf(ctx[k],     wrow[k],     acc);
    acc = fmaf(ctx[k + 1], wrow[k + 1], acc);
    acc = fmaf(ctx[k + 2], wrow[k + 2], acc);
    acc = fmaf(ctx[k + 3], wrow[k + 3], acc);
  }
  for (int qi = 0; qi < Qc; ++qi) {
    float v = queries[(size_t)qi * Dc + d] + acc;
    size_t o = (size_t)(b * Qc + qi) * Dc + d;
    qf[o] = v;
    qbf[o] = (__bf16)v;
  }
}

// K1b: gate = sigmoid(context @ gate_w^T + gate_b)  [B,Q] -> [1024]
__global__ void k_gate(const float* __restrict__ context,
                       const float* __restrict__ gate_w,
                       const float* __restrict__ gate_b,
                       float* __restrict__ gate) {
  int gid = blockIdx.x * 256 + threadIdx.x; // [0,1024)
  int b = gid >> 6, qq = gid & 63;
  const float* ctx  = context + (size_t)b * Dc;
  const float* wrow = gate_w + (size_t)qq * Dc;
  float acc = gate_b[qq];
  for (int k = 0; k < Dc; ++k) acc = fmaf(ctx[k], wrow[k], acc);
  gate[gid] = 1.f / (1.f + __expf(-acc));
}

// ---------------------------------------------------------------------------
// K2: pq = q @ qw^T + qb  (WMMA GEMM [1024 x 1024 x 1024], bf16 out)
// grid (4 col-tiles of 256, 16 row-tiles of 64), 256 threads = 8 waves.
// ---------------------------------------------------------------------------
__global__ void __launch_bounds__(256)
k_pq(const __bf16* __restrict__ qbf, const __bf16* __restrict__ qw,
     const float* __restrict__ in_b, __bf16* __restrict__ pq) {
  const int rBase = blockIdx.y * 64;
  const int cBase = blockIdx.x * 256;
  const int wv = threadIdx.x >> 5, ln = threadIdx.x & 31;
  __shared__ __bf16 As[64][264];
  v8f acc[2][4];
#pragma unroll
  for (int i = 0; i < 2; ++i)
#pragma unroll
    for (int m = 0; m < 4; ++m) acc[i][m] = vzero8();

  for (int kc = 0; kc < 4; ++kc) {
    // async global->LDS: 64x256 bf16 chunk = 2048 x 16B segments
    for (int it = 0; it < 8; ++it) {
      int e = threadIdx.x + it * 256; // [0,2048)
      int r = e >> 5, seg = e & 31;
      async_copy_b128(&As[r][seg * 8],
                      qbf + (size_t)(rBase + r) * Dc + kc * 256 + seg * 8);
    }
    wait_async0();
    __syncthreads();
#pragma unroll
    for (int ks = 0; ks < 8; ++ks) {
      const int kB = ks * 32;
      v16bf a[4];
#pragma unroll
      for (int m = 0; m < 4; ++m) a[m] = load_a_frag(&As[0][0], 264, m * 16, kB, ln);
#pragma unroll
      for (int i = 0; i < 2; ++i) {
        const int nB = cBase + (wv * 2 + i) * 16;
        v16bf b = load_b_frag(qw, Dc, nB, kc * 256 + kB, ln);
#pragma unroll
        for (int m = 0; m < 4; ++m) acc[i][m] = wmma_bf16(a[m], b, acc[i][m]);
      }
    }
    __syncthreads();
  }
  const int rOff = (ln & 16) ? 8 : 0;
#pragma unroll
  for (int i = 0; i < 2; ++i) {
    const int col = cBase + (wv * 2 + i) * 16 + (ln & 15);
    const float bias = in_b[col]; // qb
#pragma unroll
    for (int m = 0; m < 4; ++m)
#pragma unroll
      for (int v = 0; v < 8; ++v) {
        const int row = rBase + m * 16 + rOff + v;
        pq[(size_t)row * Dc + col] = (__bf16)(acc[i][m][v] + bias);
      }
  }
}

// ---------------------------------------------------------------------------
// K3: per-head kq[h][r][k] = sum_hc pq[r][h*64+hc] * kw[h*64+hc][k]
// grid (4 col-tiles, 16 row-tiles, 16 heads). K dim = 64 -> 2 WMMA steps.
// ---------------------------------------------------------------------------
__global__ void __launch_bounds__(256)
k_kq(const __bf16* __restrict__ pq, const __bf16* __restrict__ kwT,
     __bf16* __restrict__ kq) {
  const int h = blockIdx.z;
  const int rBase = blockIdx.y * 64;
  const int cBase = blockIdx.x * 256;
  const int wv = threadIdx.x >> 5, ln = threadIdx.x & 31;
  __shared__ __bf16 As[64][72];
  // async global->LDS: 64x64 bf16 = 512 x 16B segments
  for (int it = 0; it < 2; ++it) {
    int e = threadIdx.x + it * 256; // [0,512)
    int r = e >> 3, seg = e & 7;
    async_copy_b128(&As[r][seg * 8],
                    pq + (size_t)(rBase + r) * Dc + h * 64 + seg * 8);
  }
  wait_async0();
  __syncthreads();
  v8f acc[2][4];
#pragma unroll
  for (int i = 0; i < 2; ++i)
#pragma unroll
    for (int m = 0; m < 4; ++m) acc[i][m] = vzero8();
#pragma unroll
  for (int ks = 0; ks < 2; ++ks) {
    const int kB = ks * 32;
    v16bf a[4];
#pragma unroll
    for (int m = 0; m < 4; ++m) a[m] = load_a_frag(&As[0][0], 72, m * 16, kB, ln);
#pragma unroll
    for (int i = 0; i < 2; ++i) {
      const int nB = cBase + (wv * 2 + i) * 16;
      v16bf b = load_b_frag(kwT, Dc, nB, h * 64 + kB, ln);
#pragma unroll
      for (int m = 0; m < 4; ++m) acc[i][m] = wmma_bf16(a[m], b, acc[i][m]);
    }
  }
  const int rOff = (ln & 16) ? 8 : 0;
#pragma unroll
  for (int i = 0; i < 2; ++i) {
    const int col = cBase + (wv * 2 + i) * 16 + (ln & 15);
#pragma unroll
    for (int m = 0; m < 4; ++m)
#pragma unroll
      for (int v = 0; v < 8; ++v) {
        const int row = rBase + m * 16 + rOff + v;
        kq[((size_t)h * RQ + row) * Dc + col] = (__bf16)acc[i][m][v];
      }
  }
}

// ---------------------------------------------------------------------------
// K4: HBM-bound middle. Per (b,qi): stream memory tile [64 x 1024] ONCE into
// resident LDS (bf16, ~132 KB of CDNA5's 320 KB), logits_h[s] =
// 0.125*mem[s]·kq_h, masked softmax per head, then wm[h][k] = sum_s w_h[s]*
// mem[s][k] straight from resident LDS. kq chunks arrive via async-LDS.
// ---------------------------------------------------------------------------
__global__ void __launch_bounds__(256)
k_mid(const float* __restrict__ memory, const unsigned char* __restrict__ mmask,
      const __bf16* __restrict__ kq, __bf16* __restrict__ wm,
      float* __restrict__ sumw_ws) {
  const int row = blockIdx.x; // b*Q + qi
  const int b = row >> 6, qi = row & 63;
  const int tid = threadIdx.x;
  __shared__ __bf16 memA[64][1032]; // full tile resident: 132,096 B
  __shared__ __bf16 kqs[16][264];   // kq chunk: 8,448 B
  __shared__ float wrow[16][64];
  __shared__ float red[16][16];

  const int h = tid >> 4;  // head group of 16 threads
  const int sg = tid & 15;
  const int s0 = sg * 4;   // 4 slots per thread
  const size_t memBase = ((size_t)b * Mc + (size_t)qi * 64) * Dc;

  float acc[4] = {0.f, 0.f, 0.f, 0.f};
  for (int kc = 0; kc < 4; ++kc) {
    // fp32 -> bf16 convert into resident tile (VALU path; async can't convert)
    for (int it = 0; it < 16; ++it) {
      int e = tid + it * 256;
      int r = e >> 6, c4 = (e & 63) * 4;
      const float* src = memory + memBase + (size_t)r * Dc + kc * 256 + c4;
      v4bf d;
      d[0] = (__bf16)src[0]; d[1] = (__bf16)src[1];
      d[2] = (__bf16)src[2]; d[3] = (__bf16)src[3];
      *(v4bf*)&memA[r][kc * 256 + c4] = d;
    }
    // async kq chunk: 16x256 bf16 = 512 x 16B segments
    for (int it = 0; it < 2; ++it) {
      int e = tid + it * 256; // [0,512)
      int r = e >> 5, seg = e & 31;
      async_copy_b128(&kqs[r][seg * 8],
                      kq + ((size_t)r * RQ + row) * Dc + kc * 256 + seg * 8);
    }
    wait_async0();
    __syncthreads();
    for (int k = 0; k < 256; k += 8) {
      v8bf kv = *(const v8bf*)&kqs[h][k];
      float kf[8];
#pragma unroll
      for (int u = 0; u < 8; ++u) kf[u] = (float)kv[u];
#pragma unroll
      for (int j = 0; j < 4; ++j) {
        v8bf mv = *(const v8bf*)&memA[s0 + j][kc * 256 + k];
#pragma unroll
        for (int u = 0; u < 8; ++u) acc[j] = fmaf((float)mv[u], kf[u], acc[j]);
      }
    }
    __syncthreads(); // before next chunk overwrites kqs
  }

  // masked softmax over the 64 slots of this query, per head
  const float NEGINF = -3.0e38f;
  unsigned char vmask[4];
  float lg[4];
#pragma unroll
  for (int j = 0; j < 4; ++j) {
    vmask[j] = mmask[(size_t)b * Mc + qi * 64 + s0 + j];
    lg[j] = vmask[j] ? acc[j] * 0.125f : NEGINF;
  }
  red[h][sg] = fmaxf(fmaxf(lg[0], lg[1]), fmaxf(lg[2], lg[3]));
  __syncthreads();
  float m = NEGINF;
  for (int j = 0; j < 16; ++j) m = fmaxf(m, red[h][j]);
  __syncthreads();
  const bool any = (m > -1.0e37f);
  float p[4]; float ps = 0.f;
#pragma unroll
  for (int j = 0; j < 4; ++j) {
    p[j] = (any && vmask[j]) ? __expf(lg[j] - m) : 0.f;
    ps += p[j];
  }
  red[h][sg] = ps;
  __syncthreads();
  float s = 0.f;
  for (int j = 0; j < 16; ++j) s += red[h][j];
  const float inv = (s > 0.f) ? 1.f / s : 0.f;
#pragma unroll
  for (int j = 0; j < 4; ++j) wrow[h][s0 + j] = p[j] * inv;
  if (sg == 0) sumw_ws[(size_t)row * Hc + h] = (s > 0.f) ? 1.f : 0.f;
  __syncthreads();

  // wm pass straight from the resident tile: 16 heads x 1024 k, v8bf reads
  for (int it = 0; it < 8; ++it) {
    int o8 = tid + it * 256;      // [0,2048)
    int hh = o8 >> 7;             // head
    int k8 = (o8 & 127) * 8;      // k block of 8
    float a[8];
#pragma unroll
    for (int u = 0; u < 8; ++u) a[u] = 0.f;
    for (int sIdx = 0; sIdx < 64; ++sIdx) {
      const float w = wrow[hh][sIdx];
      v8bf mv = *(const v8bf*)&memA[sIdx][k8];
#pragma unroll
      for (int u = 0; u < 8; ++u) a[u] = fmaf(w, (float)mv[u], a[u]);
    }
    v8bf ov;
#pragma unroll
    for (int u = 0; u < 8; ++u) ov[u] = (__bf16)a[u];
    *(v8bf*)(wm + ((size_t)hh * RQ + row) * Dc + k8) = ov;
  }
}

// ---------------------------------------------------------------------------
// K5: per-head AV: attn[r][h*64+j] = sum_k wm[h][r][k]*vw[h*64+j][k]
//                                    + vb[h*64+j]*sumw[r][h]
// grid (16 row-tiles, 16 heads); tile 64 x 64, K = 1024 chunked by 256.
// ---------------------------------------------------------------------------
__global__ void __launch_bounds__(256)
k_av(const __bf16* __restrict__ wm, const __bf16* __restrict__ vw,
     const float* __restrict__ in_b, const float* __restrict__ sumw,
     __bf16* __restrict__ attn) {
  const int h = blockIdx.y;
  const int rBase = blockIdx.x * 64;
  const int wv = threadIdx.x >> 5, ln = threadIdx.x & 31;
  __shared__ __bf16 As[64][264];
  v8f acc[2];
  acc[0] = vzero8(); acc[1] = vzero8();
  const __bf16* vwh = vw + (size_t)h * 64 * Dc;

  for (int kc = 0; kc < 4; ++kc) {
    for (int it = 0; it < 8; ++it) {
      int e = threadIdx.x + it * 256; // [0,2048)
      int r = e >> 5, seg = e & 31;
      async_copy_b128(&As[r][seg * 8],
                      wm + ((size_t)h * RQ + rBase + r) * Dc + kc * 256 + seg * 8);
    }
    wait_async0();
    __syncthreads();
#pragma unroll
    for (int ks = 0; ks < 8; ++ks) {
      const int kB = ks * 32;
#pragma unroll
      for (int i = 0; i < 2; ++i) {
        const int t = wv * 2 + i;
        const int mt = t >> 2, nt = t & 3;
        v16bf a = load_a_frag(&As[0][0], 264, mt * 16, kB, ln);
        v16bf bfrag = load_b_frag(vwh, Dc, nt * 16, kc * 256 + kB, ln);
        acc[i] = wmma_bf16(a, bfrag, acc[i]);
      }
    }
    __syncthreads();
  }
  const int rOff = (ln & 16) ? 8 : 0;
#pragma unroll
  for (int i = 0; i < 2; ++i) {
    const int t = wv * 2 + i;
    const int mt = t >> 2, nt = t & 3;
    const int d = h * 64 + nt * 16 + (ln & 15);
    const float vb = in_b[2 * Dc + d];
#pragma unroll
    for (int v = 0; v < 8; ++v) {
      const int row = rBase + mt * 16 + rOff + v;
      attn[(size_t)row * Dc + d] =
          (__bf16)(acc[i][v] + vb * sumw[(size_t)row * Hc + h]);
    }
  }
}

// ---------------------------------------------------------------------------
// K6: out-projection WMMA GEMM + residual: ro = attn@ow^T + out_b + 0.1*q
// grid 16 row-tiles; each wave owns 8 col-tiles x 4 row-tiles.
// ---------------------------------------------------------------------------
__global__ void __launch_bounds__(256)
k_out(const __bf16* __restrict__ attn, const __bf16* __restrict__ ow,
      const float* __restrict__ ob, const float* __restrict__ qf,
      float* __restrict__ ro) {
  const int rBase = blockIdx.x * 64;
  const int wv = threadIdx.x >> 5, ln = threadIdx.x & 31;
  __shared__ __bf16 As[64][264];
  v8f acc[8][4];
#pragma unroll
  for (int nt = 0; nt < 8; ++nt)
#pragma unroll
    for (int m = 0; m < 4; ++m) acc[nt][m] = vzero8();

  for (int kc = 0; kc < 4; ++kc) {
    for (int it = 0; it < 8; ++it) {
      int e = threadIdx.x + it * 256; // [0,2048)
      int r = e >> 5, seg = e & 31;
      async_copy_b128(&As[r][seg * 8],
                      attn + (size_t)(rBase + r) * Dc + kc * 256 + seg * 8);
    }
    wait_async0();
    __syncthreads();
#pragma unroll
    for (int ks = 0; ks < 8; ++ks) {
      const int kB = ks * 32;
      v16bf a[4];
#pragma unroll
      for (int m = 0; m < 4; ++m) a[m] = load_a_frag(&As[0][0], 264, m * 16, kB, ln);
#pragma unroll
      for (int nt = 0; nt < 8; ++nt) {
        const int nB = (wv * 8 + nt) * 16;
        v16bf b = load_b_frag(ow, Dc, nB, kc * 256 + kB, ln);
#pragma unroll
        for (int m = 0; m < 4; ++m) acc[nt][m] = wmma_bf16(a[m], b, acc[nt][m]);
      }
    }
    __syncthreads();
  }
  const int rOff = (ln & 16) ? 8 : 0;
#pragma unroll
  for (int nt = 0; nt < 8; ++nt) {
    const int col = (wv * 8 + nt) * 16 + (ln & 15);
    const float bias = ob[col];
#pragma unroll
    for (int m = 0; m < 4; ++m)
#pragma unroll
      for (int v = 0; v < 8; ++v) {
        const int row = rBase + m * 16 + rOff + v;
        const size_t o = (size_t)row * Dc + col;
        ro[o] = acc[nt][m][v] + bias + 0.1f * qf[o];
      }
  }
}

// ---------------------------------------------------------------------------
// K7: LayerNorm + gate. One row per block, wave32 shuffle reductions.
// ---------------------------------------------------------------------------
__global__ void __launch_bounds__(256)
k_ln(const float* __restrict__ ro, const float* __restrict__ ln_g,
     const float* __restrict__ ln_b, const float* __restrict__ gate,
     float* __restrict__ out) {
  const int row = blockIdx.x;
  const int tid = threadIdx.x;
  __shared__ float a1[8], a2[8];
  const float* src = ro + (size_t)row * Dc;
  const int c = tid * 4;
  float x0 = src[c], x1 = src[c + 1], x2 = src[c + 2], x3 = src[c + 3];
  float s1 = x0 + x1 + x2 + x3;
  float s2 = x0 * x0 + x1 * x1 + x2 * x2 + x3 * x3;
  for (int off = 16; off > 0; off >>= 1) {
    s1 += __shfl_xor(s1, off, 32);
    s2 += __shfl_xor(s2, off, 32);
  }
  if ((tid & 31) == 0) { a1[tid >> 5] = s1; a2[tid >> 5] = s2; }
  __syncthreads();
  float S1 = 0.f, S2 = 0.f;
  for (int w = 0; w < 8; ++w) { S1 += a1[w]; S2 += a2[w]; }
  const float mu = S1 * (1.f / Dc);
  const float var = S2 * (1.f / Dc) - mu * mu;
  const float rs = rsqrtf(var + 1e-5f);
  const float g = gate[row];
  float* dst = out + (size_t)row * Dc;
  dst[c]     = ((x0 - mu) * rs * ln_g[c]     + ln_b[c])     * g;
  dst[c + 1] = ((x1 - mu) * rs * ln_g[c + 1] + ln_b[c + 1]) * g;
  dst[c + 2] = ((x2 - mu) * rs * ln_g[c + 2] + ln_b[c + 2]) * g;
  dst[c + 3] = ((x3 - mu) * rs * ln_g[c + 3] + ln_b[c + 3]) * g;
}

// ---------------------------------------------------------------------------
extern "C" void kernel_launch(void* const* d_in, const int* in_sizes, int n_in,
                              void* d_out, int out_size, void* d_ws,
                              size_t ws_size, hipStream_t stream) {
  (void)in_sizes; (void)n_in; (void)out_size; (void)ws_size;
  const float* memory  = (const float*)d_in[0];
  const float* context = (const float*)d_in[1];
  const unsigned char* mmask = (const unsigned char*)d_in[2];
  const float* queries = (const float*)d_in[3];
  const float* in_w    = (const float*)d_in[4];
  const float* in_b    = (const float*)d_in[5];
  const float* out_w   = (const float*)d_in[6];
  const float* out_b   = (const float*)d_in[7];
  const float* ctx_w   = (const float*)d_in[8];
  const float* ctx_b   = (const float*)d_in[9];
  const float* ln_g    = (const float*)d_in[10];
  const float* ln_bv   = (const float*)d_in[11];
  const float* gate_w  = (const float*)d_in[12];
  const float* gate_b  = (const float*)d_in[13];

  char* ws = (char*)d_ws;
  size_t off = 0;
  auto take = [&](size_t bytes) -> void* {
    void* p = ws + off;
    off += (bytes + 255) & ~(size_t)255;
    return p;
  };
  const size_t MB1 = (size_t)1024 * 1024;
  __bf16* qw_bf   = (__bf16*)take(MB1 * 2);
  __bf16* kwT_bf  = (__bf16*)take(MB1 * 2);
  __bf16* vw_bf   = (__bf16*)take(MB1 * 2);
  __bf16* ow_bf   = (__bf16*)take(MB1 * 2);
  float*  q_f32   = (float*)take(MB1 * 4);
  __bf16* q_bf    = (__bf16*)take(MB1 * 2);
  float*  gate_ws = (float*)take(1024 * 4);
  __bf16* pq_bf   = (__bf16*)take(MB1 * 2);
  __bf16* kq_bf   = (__bf16*)take((size_t)Hc * MB1 * 2);
  float*  sumw_ws = (float*)take((size_t)RQ * Hc * 4);
  __bf16* wm_bf   = (__bf16*)take((size_t)Hc * MB1 * 2);
  __bf16* attn_bf = (__bf16*)take(MB1 * 2);
  float*  ro_ws   = (float*)take(MB1 * 4);

  k_wcvt<<<16384, 256, 0, stream>>>(in_w, out_w, qw_bf, kwT_bf, vw_bf, ow_bf);
  k_condq<<<64, 256, 0, stream>>>(context, ctx_w, ctx_b, queries, q_f32, q_bf);
  k_gate<<<4, 256, 0, stream>>>(context, gate_w, gate_b, gate_ws);
  k_pq<<<dim3(4, 16), 256, 0, stream>>>(q_bf, qw_bf, in_b, pq_bf);
  k_kq<<<dim3(4, 16, 16), 256, 0, stream>>>(pq_bf, kwT_bf, kq_bf);
  k_mid<<<1024, 256, 0, stream>>>(memory, mmask, kq_bf, wm_bf, sumw_ws);
  k_av<<<dim3(16, 16), 256, 0, stream>>>(wm_bf, vw_bf, in_b, sumw_ws, attn_bf);
  k_out<<<16, 256, 0, stream>>>(attn_bf, ow_bf, out_b, q_f32, ro_ws);
  k_ln<<<1024, 256, 0, stream>>>(ro_ws, ln_g, ln_bv, gate_ws, (float*)d_out);
}
